// Sampler_65249143161208
// MI455X (gfx1250) — matrix operand order repaired and checked by reading, not compiled, not run on hardware
//
#include <hip/hip_runtime.h>
#include <hip/hip_fp16.h>

#define BATCH 32768
#define LDIM  128
#define HID   20
#define BL    (BATCH*LDIM)

typedef __attribute__((ext_vector_type(16))) _Float16 v16h;
typedef __attribute__((ext_vector_type(8)))  _Float16 h8;
typedef __attribute__((ext_vector_type(8)))  float    v8f;

union AMat { v16h v; h8 h[2]; };

// Wave-private LDS sync: same-wave LDS ops are in-order in HW (DScnt, in order);
// we only need a compiler memory barrier, zero ISA cost.
__device__ __forceinline__ void wave_sync(){
  __builtin_amdgcn_fence(__ATOMIC_RELEASE, "wavefront");
  __builtin_amdgcn_wave_barrier();
  __builtin_amdgcn_fence(__ATOMIC_ACQUIRE, "wavefront");
}

__device__ __forceinline__ unsigned pcg(unsigned v){
  unsigned s = v*747796405u + 2891336453u;
  unsigned w = ((s >> ((s>>28u)+4u)) ^ s) * 277803737u;
  return (w>>22u) ^ w;
}
__device__ __forceinline__ float nrand(unsigned n){
  unsigned a = pcg(n);
  unsigned b = pcg(n ^ 0x9E3779B9u);
  float u1 = ((float)a + 1.0f) * 2.3283064365386963e-10f; // (0,1]
  float u2 = (float)b * 2.3283064365386963e-10f;
  return __builtin_amdgcn_sqrtf(fmaxf(-2.0f*__logf(u1), 0.0f)) * __cosf(6.28318530718f*u2);
}
// v_exp_f32 + v_rcp_f32, overflow-safe at both ends, no IEEE div sequence
__device__ __forceinline__ float sigm(float x){
  return __builtin_amdgcn_rcpf(1.0f + __expf(-x));          // x->+inf: 1, x->-inf: 0
}
__device__ __forceinline__ float tanh_f(float x){
  float t = __expf(2.0f*x);                                  // inf ok
  return __builtin_fmaf(-2.0f, __builtin_amdgcn_rcpf(t + 1.0f), 1.0f); // -> [-1,1]
}
__device__ __forceinline__ float leaky(float x){ return x>0.0f ? x : 0.01f*x; }

__global__ __launch_bounds__(128,1) void sampler_lstm_kernel(
  const float* __restrict__ mu, const float* __restrict__ log_var,
  const float* __restrict__ W_ih1, const float* __restrict__ W_hh1,
  const float* __restrict__ b_ih1, const float* __restrict__ b_hh1,
  const float* __restrict__ W_ih2, const float* __restrict__ W_hh2,
  const float* __restrict__ b_ih2, const float* __restrict__ b_hh2,
  float* __restrict__ out)
{
  __shared__ float    Wsh[80*127];      // W_ih1, reused every step (rank-1 updates)
  __shared__ _Float16 hAs[4][16*32];    // per-wave h1 staging (A-layout source), ch 20..31 stay 0
  __shared__ _Float16 oAs[4][16*64];    // per-wave out1||h2 staging, cols 42..63 stay 0
  __shared__ float    smu[4][16];       // mu column broadcast
  __shared__ float    sfi[4][16];       // first_input
  __shared__ float    sfd[4][16];       // mu[:,0]-first_input (step-1 correction)
  __shared__ float    sh2[4][32];       // h2 (f32) for outputs

  const int tid  = threadIdx.x;
  const int wid  = tid >> 5;
  const int lane = tid & 31;
  const int n    = lane & 15;
  const int hi   = lane >> 4;
  const int base = (blockIdx.x*4 + wid) * 16;   // 16 batch rows per wave

  for (int i=tid; i<80*127;  i+=128) Wsh[i] = W_ih1[i];
  for (int i=tid; i<4*16*32; i+=128) (&hAs[0][0])[i] = (_Float16)0.0f;
  for (int i=tid; i<4*16*64; i+=128) (&oAs[0][0])[i] = (_Float16)0.0f;

  if (lane < 16){
    int b = base + lane;
    float mu0 = mu[b*LDIM];
    float lv0 = log_var[b*LDIM];
    float fi  = nrand((unsigned)b)*__expf(0.5f*lv0) + mu0;
    sfi[wid][lane] = fi;
    sfd[wid][lane] = mu0 - fi;
    __builtin_nontemporal_store(0.0f, &out[b*LDIM]);        // mus_full[:,0]
    __builtin_nontemporal_store(1.0f, &out[BL + b*LDIM]);   // lvs_full[:,0]
    __builtin_nontemporal_store(fi,   &out[2*BL + b*LDIM]); // samples[:,0]
  }

  // Per-lane gate-tile constants. Padded gate col layout: [i0..31|f0..31|g0..31|o0..31]
  int   wrowB[8]; float bc1v[8]; bool val1[8];
  v16h  Bh[8];                               // W_hh1^T tiles in WMMA B layout (f16)
  #pragma unroll
  for (int t=0;t<8;++t){
    int  type = t>>1;
    int  chp  = ((t&1)<<4) | n;
    bool v    = chp < HID;
    int  wrow = type*HID + chp;
    val1[t]  = v;
    wrowB[t] = v ? wrow*127 : 0;
    bc1v[t]  = v ? (b_ih1[wrow] + b_hh1[wrow]) : 0.0f;
    #pragma unroll
    for (int e=0;e<16;++e){
      int k = hi*8 + e + ((e>>3)<<3);
      float w = (v && k < HID) ? W_hh1[wrow*HID + k] : 0.0f;
      Bh[t][e] = (_Float16)w;
    }
  }
  // LSTM2 B tiles: rows 0..39 = W_ih2^T, rows 40..41 = W_hh2^T (h2 folded into K)
  v16h B20, B21;
  float bc2 = (n<8) ? (b_ih2[n] + b_hh2[n]) : 0.0f;
  #pragma unroll
  for (int e=0;e<16;++e){
    int k = hi*8 + e + ((e>>3)<<3);
    B20[e] = (n<8) ? (_Float16)W_ih2[n*40 + k] : (_Float16)0.0f;
    int kg = 32 + hi*8 + e + ((e>>3)<<3);
    float w2 = 0.0f;
    if (n<8){
      if      (kg < 40)  w2 = W_ih2[n*40 + kg];
      else if (kg == 40) w2 = W_hh2[n*2 + 0];
      else if (kg == 41) w2 = W_hh2[n*2 + 1];
    }
    B21[e] = (_Float16)w2;
  }

  __syncthreads();   // Wsh + zero-init + sfi visible block-wide (only cross-wave sync needed)

  // Accumulator (x-prefix @ W_ih1^T + biases), C/D layout, f32
  v8f acc[8];
  #pragma unroll
  for (int t=0;t<8;++t){
    float w0 = val1[t] ? Wsh[wrowB[t]] : 0.0f;
    #pragma unroll
    for (int r=0;r<8;++r) acc[t][r] = bc1v[t] + sfi[wid][r + 8*hi]*w0;
  }

  v8f c1p[2], c2v;
  #pragma unroll
  for (int r=0;r<8;++r){ c1p[0][r]=0.0f; c1p[1][r]=0.0f; c2v[r]=0.0f; }

  const int m0 = n;        // A-matrix row handled by this lane
  const int kb = hi*8;     // A-matrix K base

  // mu column prefetch register (column idx for iteration idx)
  float vmu = 0.0f;
  if (lane < 16) vmu = mu[(base+lane)*LDIM + 1];

  for (int idx=1; idx<LDIM; ++idx){
    // ---- s1: A1 = h1(prev) from LDS; publish prefetched mu column
    AMat a1;
    a1.h[0] = *(const h8*)&hAs[wid][m0*32 + kb];
    a1.h[1] = *(const h8*)&hAs[wid][m0*32 + kb + 16];
    if (lane < 16) smu[wid][lane] = vmu;
    wave_sync();
    // prefetch next column; latency hidden behind this whole step
    if (lane < 16 && idx < 126) vmu = mu[(base+lane)*LDIM + idx + 1];

    // ---- s2: gates1 = A1*Whh + acc ; elementwise LSTM1 ; stage h1/out1
    #pragma unroll
    for (int p=0;p<2;++p){
      v8f Gi = __builtin_amdgcn_wmma_f32_16x16x32_f16(false, a1.v, false, Bh[0+p], (short)0, acc[0+p], false, false);
      v8f Gf = __builtin_amdgcn_wmma_f32_16x16x32_f16(false, a1.v, false, Bh[2+p], (short)0, acc[2+p], false, false);
      v8f Gg = __builtin_amdgcn_wmma_f32_16x16x32_f16(false, a1.v, false, Bh[4+p], (short)0, acc[4+p], false, false);
      v8f Go = __builtin_amdgcn_wmma_f32_16x16x32_f16(false, a1.v, false, Bh[6+p], (short)0, acc[6+p], false, false);
      int ch = (p<<4) | n;
      #pragma unroll
      for (int r=0;r<8;++r){
        float iv = sigm(Gi[r]);
        float fv = sigm(Gf[r]);
        float gv = tanh_f(Gg[r]);
        float ov = sigm(Go[r]);
        float cn = fv*c1p[p][r] + iv*gv;
        c1p[p][r] = cn;
        float hn = ov*tanh_f(cn);
        int m = r + 8*hi;
        hAs[wid][m*32 + ch] = (_Float16)hn;
        if (ch < HID){
          oAs[wid][m*64 + ch]       = (_Float16)leaky(hn);
          oAs[wid][m*64 + 20 + ch]  = (_Float16)leaky(cn);
        }
      }
    }
    wave_sync();

    // ---- s3: LSTM2 (2 chained WMMAs, h2 recurrence folded into K=40..41) + acc rank-1 update
    AMat a20, a21;
    a20.h[0] = *(const h8*)&oAs[wid][m0*64 + kb];
    a20.h[1] = *(const h8*)&oAs[wid][m0*64 + kb + 16];
    a21.h[0] = *(const h8*)&oAs[wid][m0*64 + 32 + kb];
    a21.h[1] = *(const h8*)&oAs[wid][m0*64 + 32 + kb + 16];
    v8f G2;
    #pragma unroll
    for (int r=0;r<8;++r) G2[r] = bc2;
    G2 = __builtin_amdgcn_wmma_f32_16x16x32_f16(false, a20.v, false, B20, (short)0, G2, false, false);
    G2 = __builtin_amdgcn_wmma_f32_16x16x32_f16(false, a21.v, false, B21, (short)0, G2, false, false);
    #pragma unroll
    for (int r=0;r<8;++r){
      int fb = __builtin_amdgcn_ds_bpermute((lane+2)*4, __float_as_int(G2[r]));
      int gb = __builtin_amdgcn_ds_bpermute((lane+4)*4, __float_as_int(G2[r]));
      int ob = __builtin_amdgcn_ds_bpermute((lane+6)*4, __float_as_int(G2[r]));
      float ig = sigm(G2[r]);
      float fg = sigm(__int_as_float(fb));
      float gg = tanh_f(__int_as_float(gb));
      float og = sigm(__int_as_float(ob));
      float cn = fg*c2v[r] + ig*gg;
      c2v[r] = cn;
      float hn = og*tanh_f(cn);
      if (n < 2){
        int m = r + 8*hi;
        oAs[wid][m*64 + 40 + n] = (_Float16)hn;   // h2 for next step's K
        sh2[wid][m*2 + n] = hn;
      }
    }
    if (idx < 127){
      float muv[8];
      #pragma unroll
      for (int r=0;r<8;++r) muv[r] = smu[wid][r + 8*hi];
      #pragma unroll
      for (int t=0;t<8;++t){
        float wv = val1[t] ? Wsh[wrowB[t] + idx] : 0.0f;
        #pragma unroll
        for (int r=0;r<8;++r) acc[t][r] += muv[r]*wv;
      }
      if (idx == 1){  // replace first_input column by mu[:,0]
        float dv[8];
        #pragma unroll
        for (int r=0;r<8;++r) dv[r] = sfd[wid][r + 8*hi];
        #pragma unroll
        for (int t=0;t<8;++t){
          float w0 = val1[t] ? Wsh[wrowB[t]] : 0.0f;
          #pragma unroll
          for (int r=0;r<8;++r) acc[t][r] += dv[r]*w0;
        }
      }
    }
    wave_sync();

    // ---- s4: outputs for this step (streamed, no reuse -> nontemporal)
    if (lane < 16){
      int b = base + lane;
      float mur = sh2[wid][lane*2 + 0];
      float lvr = sh2[wid][lane*2 + 1];
      float ev  = nrand((unsigned)(idx*BATCH + b));
      float sp  = ev*__expf(0.5f*lvr) + mur;
      __builtin_nontemporal_store(mur, &out[b*LDIM + idx]);
      __builtin_nontemporal_store(lvr, &out[BL + b*LDIM + idx]);
      __builtin_nontemporal_store(sp,  &out[2*BL + b*LDIM + idx]);
    }
  }
}

extern "C" void kernel_launch(void* const* d_in, const int* in_sizes, int n_in,
                              void* d_out, int out_size, void* d_ws, size_t ws_size,
                              hipStream_t stream) {
  const float* mu     = (const float*)d_in[0];
  const float* lv     = (const float*)d_in[1];
  const float* W_ih1  = (const float*)d_in[2];
  const float* W_hh1  = (const float*)d_in[3];
  const float* b_ih1  = (const float*)d_in[4];
  const float* b_hh1  = (const float*)d_in[5];
  const float* W_ih2  = (const float*)d_in[6];
  const float* W_hh2  = (const float*)d_in[7];
  const float* b_ih2  = (const float*)d_in[8];
  const float* b_hh2  = (const float*)d_in[9];
  float* out = (float*)d_out;
  // 512 blocks x 128 threads; each block = 4 waves, each wave owns 16 batch rows
  sampler_lstm_kernel<<<BATCH/64, 128, 0, stream>>>(
      mu, lv, W_ih1, W_hh1, b_ih1, b_hh1, W_ih2, W_hh2, b_ih2, b_hh2, out);
}
